// AdvancedNeRF_64072322121976
// MI455X (gfx1250) — matrix operand order repaired and checked by reading, not compiled, not run on hardware
//
#include <hip/hip_runtime.h>

typedef __attribute__((ext_vector_type(16))) _Float16 v16h;
typedef __attribute__((ext_vector_type(8)))  float    v8f;
typedef int v4i __attribute__((vector_size(16)));     // gcc-style int4, matches builtin param

#define HID      256
#define ROWS     128          // samples per workgroup
#define NTHREADS 256
#define CHUNK_H  16384        // halves per staged chunk (32 KB = 64 weight rows @ stride 256)

// workspace layout in halves (all Wt stored as [N][K] row-major, K padded)
#define O_IN   0              // Wt_in   [256][64]   = 16384  (1 chunk)
#define O_D    16384          // Wt_d    8x[256][256]= 524288 (32 chunks)
#define O_F    540672         // Wt_feat [128][256]  = 32768  (2 chunks)
#define O_C0   573440         // Wt_c0   [128][256]  = 32768  (2 chunks, K padded 224->256)
#define WS_HALVES 606208
#define NCHUNKS 37

#if defined(__AMDGCN__) && __has_builtin(__builtin_amdgcn_global_load_async_to_lds_b128)
#define HAVE_ASYNC 1
#else
#define HAVE_ASYNC 0
#endif

// ---------------- fragment loaders ----------------
// A (16x32 f16): lanes 0-15 -> M=lane, halves = K k0..k0+15
//                lanes 16-31 -> M=lane-16, halves = K k0+16..k0+31
__device__ __forceinline__ v16h ld_afrag(const _Float16* act, int row0, int k0, int lane) {
  int m  = lane & 15;
  int kb = k0 + ((lane >> 4) << 4);
  const _Float16* p = act + (row0 + m) * HID + kb;
  v16h r;
  reinterpret_cast<float4*>(&r)[0] = *reinterpret_cast<const float4*>(p);
  reinterpret_cast<float4*>(&r)[1] = *reinterpret_cast<const float4*>(p + 8);
  return r;
}
// B (32x16 f16): lane holds column n = n0 + (lane&15); lanes 0-15 hold K k0..k0+15,
// lanes 16-31 hold K k0+16..k0+31 (2 K per VGPR, contiguous per lane)
__device__ __forceinline__ v16h ld_bfrag(const _Float16* w, int stride, int n0, int k0, int lane) {
  int n  = n0 + (lane & 15);
  int kb = k0 + ((lane >> 4) << 4);
  const _Float16* p = w + n * stride + kb;
  v16h r;
  reinterpret_cast<float4*>(&r)[0] = reinterpret_cast<const float4*>(p)[0];
  reinterpret_cast<float4*>(&r)[1] = reinterpret_cast<const float4*>(p)[1];
  return r;
}

__device__ __forceinline__ v8f wmma_f16(v16h a, v16h b, v8f c) {
  return __builtin_amdgcn_wmma_f32_16x16x32_f16(false, a, false, b, (short)0, c, false, false);
}

// ---- async (or fallback) chunk staging: 16384 halves, 8 x b128 per thread ----
__device__ __forceinline__ void issue_chunk(const _Float16* __restrict__ src,
                                            _Float16* __restrict__ dst, int tid) {
#if HAVE_ASYNC
  #pragma unroll
  for (int i = 0; i < 8; ++i) {
    int off = i * 2048 + tid * 8;
    __builtin_amdgcn_global_load_async_to_lds_b128(
        (__attribute__((address_space(1))) v4i*)(src + off),
        (__attribute__((address_space(3))) v4i*)(dst + off), 0, 0);
  }
#else
  #pragma unroll
  for (int i = 0; i < 8; ++i) {
    int off = i * 2048 + tid * 8;
    *reinterpret_cast<float4*>(dst + off) = *reinterpret_cast<const float4*>(src + off);
  }
#endif
}
__device__ __forceinline__ void wait_chunks() {
#if HAVE_ASYNC
#if __has_builtin(__builtin_amdgcn_s_wait_asynccnt)
  __builtin_amdgcn_s_wait_asynccnt(0);
#else
  asm volatile("s_wait_asynccnt 0x0" ::: "memory");
#endif
#endif
  __syncthreads();
}
__device__ __forceinline__ const _Float16* chunk_src(const _Float16* ws, int c) {
  if (c == 0)  return ws + O_IN;
  if (c <= 32) return ws + O_D  + (c - 1)  * CHUNK_H;
  if (c <= 34) return ws + O_F  + (c - 33) * CHUNK_H;
  return ws + O_C0 + (c - 35) * CHUNK_H;
}

// one wave computes ntiles 16-wide output tiles of (16 rows) x W^T, writes f16 to act
template<int KSTEPS>
__device__ __forceinline__ void gemm_tiles(const v16h* aF, const _Float16* wbuf, int stride,
                                           int ntiles, int n_glob0, const float* __restrict__ bias,
                                           bool relu, _Float16* act, int row0, int outcol0, int lane) {
  const int m0 = (lane >> 4) << 3;          // D: lanes 16-31 hold M=8..15
  const int nl = lane & 15;
  for (int t = 0; t < ntiles; ++t) {
    v8f acc = {};
    #pragma unroll
    for (int s = 0; s < KSTEPS; ++s) {
      v16h b = ld_bfrag(wbuf, stride, t * 16, s * 32, lane);
      acc = wmma_f16(aF[s], b, acc);
    }
    int n = t * 16 + nl;
    float bv = bias[n_glob0 + n];
    #pragma unroll
    for (int v = 0; v < 8; ++v) {
      float x = acc[v] + bv;
      if (relu) x = fmaxf(x, 0.f);
      act[(row0 + m0 + v) * HID + outcol0 + n] = (_Float16)x;
    }
  }
}

// ---------------- weight conversion: f32 -> f16 transposed ----------------
__global__ void convert_weights(const float* __restrict__ w_in, const float* __restrict__ dW,
                                const float* __restrict__ w_feat, const float* __restrict__ cw0,
                                _Float16* __restrict__ ws) {
  int i = blockIdx.x * 256 + threadIdx.x;
  if (i >= WS_HALVES) return;
  float v = 0.f;
  if (i < O_D) {                              // Wt_in [256][64], src w_in [63,256]
    int n = i >> 6, k = i & 63;
    if (k < 63) v = w_in[k * 256 + n];
  } else if (i < O_F) {                       // Wt_d [l][256][256], src dW [8,256,256]
    int j = i - O_D;
    int l = j >> 16, r = j & 65535;
    int n = r >> 8, k = r & 255;
    v = dW[l * 65536 + k * 256 + n];
  } else if (i < O_C0) {                      // Wt_feat [128][256], src w_feat [256,128]
    int j = i - O_F;
    int n = j >> 8, k = j & 255;
    v = w_feat[k * 128 + n];
  } else {                                    // Wt_c0 [128][256], src cw0 [203,128], K pad
    int j = i - O_C0;
    int n = j >> 8, k = j & 255;
    if (k < 203) v = cw0[k * 128 + n];
  }
  ws[i] = (_Float16)v;
}

// ---------------- fused NeRF forward ----------------
__global__ __launch_bounds__(NTHREADS) void nerf_fused(
    const float* __restrict__ pos, const float* __restrict__ dirs, const int* __restrict__ ids,
    const float* __restrict__ b_in, const float* __restrict__ dB,
    const float* __restrict__ w_sig, const float* __restrict__ b_sig,
    const float* __restrict__ b_feat, const float* __restrict__ emb,
    const float* __restrict__ cb0, const float* __restrict__ cw1, const float* __restrict__ cb1,
    const _Float16* __restrict__ ws, float* __restrict__ out, int N) {
  __shared__ _Float16 act[ROWS * HID];        // 64 KB activations (f16)
  __shared__ _Float16 wbuf[2][CHUNK_H];       // 2 x 32 KB double-buffered weights

  const int tid  = threadIdx.x;
  const int lane = tid & 31;
  const int wave = tid >> 5;
  const int row0 = wave * 16;                 // this wave's 16 sample rows
  const int s0   = blockIdx.x * ROWS;

  issue_chunk(chunk_src(ws, 0), wbuf[0], tid);   // chunk 0 (input weights) in flight

  // ---- positional encoding -> act[r][0..63] (overlaps chunk 0 DMA) ----
  if (tid < ROWS) {
    int s = s0 + tid;
    float x = 0.f, y = 0.f, z = 0.f;
    if (s < N) { x = pos[s * 3 + 0]; y = pos[s * 3 + 1]; z = pos[s * 3 + 2]; }
    _Float16* a = act + tid * HID;
    a[0] = (_Float16)x; a[1] = (_Float16)y; a[2] = (_Float16)z;
    float v3[3] = {x, y, z};
    for (int c = 0; c < 3; ++c) {
      float f = 1.f;
      for (int j = 0; j < 10; ++j) {
        float sn, cs; __sincosf(v3[c] * f, &sn, &cs);
        a[3 + c * 20 + j]      = (_Float16)sn;
        a[3 + c * 20 + 10 + j] = (_Float16)cs;
        f *= 2.f;
      }
    }
    a[63] = (_Float16)0.f;
  }
  wait_chunks();                                 // chunk 0 ready, posenc visible
  issue_chunk(chunk_src(ws, 1), wbuf[1], tid);   // density l0 q0 streams during input layer

  // ---- input layer: [16,64] @ [64,256], ReLU ----
  {
    v16h aF[2];
    aF[0] = ld_afrag(act, row0, 0, lane);
    aF[1] = ld_afrag(act, row0, 32, lane);
    gemm_tiles<2>(aF, wbuf[0], 64, 16, 0, b_in, true, act, row0, 0, lane);
  }

  // ---- 8 density layers [16,256] @ [256,256]; ReLU on first 7; 4 chunks each ----
  for (int l = 0; l < 8; ++l) {
    v16h aF[8];
    bool relu = (l < 7);
    for (int q = 0; q < 4; ++q) {
      int c = 1 + l * 4 + q;
      wait_chunks();                             // chunk c landed; prev compute done WG-wide
      if (q == 0) {
        #pragma unroll
        for (int s = 0; s < 8; ++s) aF[s] = ld_afrag(act, row0, s * 32, lane);
      }
      issue_chunk(chunk_src(ws, c + 1), wbuf[(c + 1) & 1], tid);  // prefetch next
      gemm_tiles<8>(aF, wbuf[c & 1], 256, 4, q * 64, dB + l * 256, relu,
                    act, row0, q * 64, lane);
    }
  }

  wait_chunks();                                 // chunk 33 (feat rows 0..63); h complete
  issue_chunk(chunk_src(ws, 34), wbuf[0], tid);

  // preload h for the feature matmul before act gets overwritten
  v16h aH[8];
  #pragma unroll
  for (int s = 0; s < 8; ++s) aH[s] = ld_afrag(act, row0, s * 32, lane);

  // ---- density head: softplus(h @ w_sig + b_sig) ----
  if (tid < ROWS) {
    int s = s0 + tid;
    if (s < N) {
      float acc = b_sig[0];
      const _Float16* a = act + tid * HID;
      for (int k = 0; k < HID; ++k) acc += (float)a[k] * w_sig[k];
      out[(size_t)N * 3 + s] = (acc > 20.f) ? acc : log1pf(__expf(acc));
    }
  }
  __syncthreads();                               // all cross-wave reads of h done

  // ---- direction encoding + appearance embedding -> act[r][128..223] ----
  if (tid < ROWS) {
    int s = s0 + tid;
    _Float16* a = act + tid * HID;
    float dx = 0.f, dy = 0.f, dz = 0.f; int id = 0;
    if (s < N) { dx = dirs[s * 3 + 0]; dy = dirs[s * 3 + 1]; dz = dirs[s * 3 + 2]; id = ids[s]; }
    a[128] = (_Float16)dx; a[129] = (_Float16)dy; a[130] = (_Float16)dz;
    float v3[3] = {dx, dy, dz};
    for (int c = 0; c < 3; ++c) {
      float f = 1.f;
      for (int j = 0; j < 4; ++j) {
        float sn, cs; __sincosf(v3[c] * f, &sn, &cs);
        a[131 + c * 8 + j]     = (_Float16)sn;
        a[131 + c * 8 + 4 + j] = (_Float16)cs;
        f *= 2.f;
      }
    }
    const float* e = emb + (size_t)id * 48;
    for (int j = 0; j < 48; ++j) a[155 + j] = (_Float16)e[j];
    for (int j = 203; j < 224; ++j) a[j] = (_Float16)0.f;   // pad K 203->224
  }
  // ---- color features q0: h @ w_feat (no ReLU) -> act[r][0..63] ----
  gemm_tiles<8>(aH, wbuf[1], 256, 4, 0, b_feat, false, act, row0, 0, lane);

  wait_chunks();                                 // chunk 34; feat q0 + direnc done WG-wide
  issue_chunk(chunk_src(ws, 35), wbuf[1], tid);
  // ---- color features q1 -> act[r][64..127] ----
  gemm_tiles<8>(aH, wbuf[0], 256, 4, 64, b_feat, false, act, row0, 64, lane);

  wait_chunks();                                 // chunk 35 (cw0 rows 0..63); feat q1 done
  issue_chunk(chunk_src(ws, 36), wbuf[0], tid);
  v16h aC[7];
  #pragma unroll
  for (int s = 0; s < 7; ++s) aC[s] = ld_afrag(act, row0, s * 32, lane);
  // ---- color layer q0: relu(color_in @ cw0 + cb0) -> act[r][0..63] ----
  gemm_tiles<7>(aC, wbuf[1], 256, 4, 0, cb0, true, act, row0, 0, lane);

  wait_chunks();                                 // chunk 36; color q0 done
  // ---- color layer q1 -> act[r][64..127] ----
  gemm_tiles<7>(aC, wbuf[0], 256, 4, 64, cb0, true, act, row0, 64, lane);
  __syncthreads();

  // ---- color head: sigmoid(c @ cw1 + cb1) ----
  if (tid < ROWS) {
    int s = s0 + tid;
    if (s < N) {
      const _Float16* a = act + tid * HID;
      float c0 = cb1[0], c1 = cb1[1], c2 = cb1[2];
      for (int k = 0; k < 128; ++k) {
        float av = (float)a[k];
        c0 += av * cw1[k * 3 + 0];
        c1 += av * cw1[k * 3 + 1];
        c2 += av * cw1[k * 3 + 2];
      }
      out[(size_t)s * 3 + 0] = 1.f / (1.f + __expf(-c0));
      out[(size_t)s * 3 + 1] = 1.f / (1.f + __expf(-c1));
      out[(size_t)s * 3 + 2] = 1.f / (1.f + __expf(-c2));
    }
  }
}

extern "C" void kernel_launch(void* const* d_in, const int* in_sizes, int n_in,
                              void* d_out, int out_size, void* d_ws, size_t ws_size,
                              hipStream_t stream) {
  const float* pos    = (const float*)d_in[0];
  const float* dirs   = (const float*)d_in[1];
  const int*   ids    = (const int*)d_in[2];
  const float* w_in   = (const float*)d_in[3];
  const float* b_in   = (const float*)d_in[4];
  const float* dW     = (const float*)d_in[5];
  const float* dB     = (const float*)d_in[6];
  const float* w_sig  = (const float*)d_in[7];
  const float* b_sig  = (const float*)d_in[8];
  const float* w_feat = (const float*)d_in[9];
  const float* b_feat = (const float*)d_in[10];
  const float* emb    = (const float*)d_in[11];
  const float* cw0    = (const float*)d_in[12];
  const float* cb0    = (const float*)d_in[13];
  const float* cw1    = (const float*)d_in[14];
  const float* cb1    = (const float*)d_in[15];
  _Float16* ws  = (_Float16*)d_ws;
  float*    out = (float*)d_out;
  int N = in_sizes[0] / 3;

  convert_weights<<<(WS_HALVES + 255) / 256, 256, 0, stream>>>(w_in, dW, w_feat, cw0, ws);
  int blocks = (N + ROWS - 1) / ROWS;
  nerf_fused<<<blocks, NTHREADS, 0, stream>>>(pos, dirs, ids, b_in, dB, w_sig, b_sig,
                                              b_feat, emb, cb0, cw1, cb1, ws, out, N);
}
